// _LLCMBlock_67688684585101
// MI455X (gfx1250) — matrix operand order repaired and checked by reading, compiled, not run on hardware
//
#include <hip/hip_runtime.h>
#include <hip/hip_bf16.h>

// ---------------------------------------------------------------------------
// Lorentz-attention transformer block for gfx1250 (MI455X), bf16 WMMA path.
//   D=128, H=4, DH=32, TD=8 (per-head time dims), S=2048, B=8, DFF=512
// All GEMMs run on v_wmma_f32_16x16x32_bf16 (f32 accumulate).
// eta (Minkowski metric) and 1/sqrt(DH) are folded into Wq / bq.
// Attention is single-pass flash (online softmax); softmax row reductions use
// v_permlane16_b32 butterflies (VALU) instead of ds_bpermute.
// ---------------------------------------------------------------------------

typedef __bf16 bf16;
typedef __attribute__((ext_vector_type(16))) __bf16 v16bf;
typedef __attribute__((ext_vector_type(8)))  float  v8f;

#define DMODEL 128
#define NHEAD  4
#define DHEAD  32
#define SEQ    2048
#define BATCH  8
#define MTOK   (BATCH * SEQ)   // 16384 tokens
#define DFF    512
#define KCHUNK 64              // keys staged per LDS tile

__device__ __forceinline__ bf16 f2bf(float f) { return (bf16)f; }

union AB16 { v16bf v; float4 f4[2]; };   // 32-byte fragment as two b128 loads
union V8B  { float4 f4; bf16 e[8]; };

// ---- 16-lane xor-butterfly via v_permlane16_b32 (no LDS traffic) ----------
template <unsigned LO, unsigned HI>
__device__ __forceinline__ float plane16(float x) {
  int i = __float_as_int(x);
  int r = __builtin_amdgcn_permlane16(i, i, (int)LO, (int)HI, false, false);
  return __int_as_float(r);
}
__device__ __forceinline__ float rowmax16(float x) {
  x = fmaxf(x, plane16<0x67452301u, 0xEFCDAB89u>(x));   // xor 1
  x = fmaxf(x, plane16<0x54761032u, 0xDCFEBA98u>(x));   // xor 2 on lanes 0..7, 12..15; lanes 8..11 read themselves
  x = fmaxf(x, plane16<0x32107654u, 0xBA98FEDCu>(x));   // xor 4
  x = fmaxf(x, plane16<0xFEDCBA98u, 0x76543210u>(x));   // xor 8
  return x;
}
__device__ __forceinline__ float rowsum16(float x) {
  x += plane16<0x67452301u, 0xEFCDAB89u>(x);
  x += plane16<0x54761032u, 0xDCFEBA98u>(x);
  x += plane16<0x32107654u, 0xBA98FEDCu>(x);
  x += plane16<0xFEDCBA98u, 0x76543210u>(x);
  return x;
}

// ---------------------------------------------------------------------------
// Generic wave-tile GEMM: C[M,N] = A[M,K](bf16) @ Bt[N,K](bf16)^T + bias
// MODE 0: store bf16   MODE 1: +res (f32), store f32   MODE 2: GELU, store bf16
// ---------------------------------------------------------------------------
template <int MODE>
__global__ void gemm16(const bf16* __restrict__ A, const bf16* __restrict__ Bt,
                       const float* __restrict__ bias, const float* __restrict__ res,
                       void* __restrict__ Cout, int M, int N, int K)
{
  int wid  = blockIdx.x * (blockDim.x >> 5) + (threadIdx.x >> 5);
  int lane = threadIdx.x & 31;
  int ntiles = N >> 4;
  int tm = wid / ntiles, tn = wid % ntiles;
  if (tm * 16 >= M) return;
  int r = lane & 15, g = lane >> 4;

  const bf16* arow = A  + (size_t)(tm * 16 + r) * K + 8 * g;
  const bf16* brow = Bt + (size_t)(tn * 16 + r) * K + 8 * g;

  v8f acc = {};
  for (int k0 = 0; k0 < K; k0 += 32) {
    AB16 a, b;
    a.f4[0] = *(const float4*)(arow + k0);
    a.f4[1] = *(const float4*)(arow + k0 + 16);
    b.f4[0] = *(const float4*)(brow + k0);
    b.f4[1] = *(const float4*)(brow + k0 + 16);
    acc = __builtin_amdgcn_wmma_f32_16x16x32_bf16(false, a.v, false, b.v,
                                                  (short)0, acc, false, false);
  }

  int n = tn * 16 + r;
  float bn = bias ? bias[n] : 0.0f;
#pragma unroll
  for (int i = 0; i < 8; ++i) {
    int m = tm * 16 + i + 8 * g;
    float v = acc[i] + bn;
    if constexpr (MODE == 1) {
      ((float*)Cout)[(size_t)m * N + n] = v + res[(size_t)m * N + n];
    } else if constexpr (MODE == 2) {
      float gl = 0.5f * v * (1.0f + erff(v * 0.70710678118f));  // exact GELU
      ((bf16*)Cout)[(size_t)m * N + n] = f2bf(gl);
    } else {
      ((bf16*)Cout)[(size_t)m * N + n] = f2bf(v);
    }
  }
}

// ---------------------------------------------------------------------------
// Flash attention, eta/scale pre-folded into Q.
// Block = 256 thr (8 waves); each wave owns 16 query rows; block owns 128 rows
// of one (batch, head). Keys streamed in 64-wide chunks through LDS.
// Per chunk per wave: 4 WMMA for QK^T, 4 WMMA for PV; one max/sum permlane tree.
// ---------------------------------------------------------------------------
__global__ void lorentz_flash_attn(const bf16* __restrict__ Q,
                                   const bf16* __restrict__ Kg,
                                   const bf16* __restrict__ Vg,
                                   bf16* __restrict__ O)
{
  __shared__ alignas(16) bf16 sK [KCHUNK * 32];        // [key][d]
  __shared__ alignas(16) bf16 sVt[32 * KCHUNK];        // [d][key] (transposed)
  __shared__ alignas(16) bf16 sP [8][16 * KCHUNK];     // per-wave P staging

  int tid = threadIdx.x, lane = tid & 31, w = tid >> 5;
  int qb = blockIdx.x & 15;            // 16 query blocks of 128
  int bh = blockIdx.x >> 4;
  int h = bh & (NHEAD - 1), b = bh / NHEAD;
  int r = lane & 15, g = lane >> 4;

  const size_t rowbase = (size_t)b * SEQ;
  int q0 = qb * 128 + w * 16;

  // Q A-fragment (loop-invariant); eta & 1/sqrt(DH) already folded into Q.
  AB16 qa;
  const bf16* qrow = Q + (rowbase + q0 + r) * DMODEL + h * DHEAD + 8 * g;
  qa.f4[0] = *(const float4*)(qrow);
  qa.f4[1] = *(const float4*)(qrow + 16);

  float mrow[8], lrow[8];
  v8f acc0 = {}, acc1 = {};
#pragma unroll
  for (int i = 0; i < 8; ++i) { mrow[i] = -3.0e38f; lrow[i] = 0.0f; }

  // cooperative K/V staging: 256 threads x 8 bf16 each per tensor (64x32 tile)
  int kk = tid >> 2;            // key 0..63
  int dg = (tid & 3) * 8;       // dim group (8 contiguous dims)
  const bf16* kbaseg = Kg + rowbase * DMODEL + h * DHEAD;
  const bf16* vbaseg = Vg + rowbase * DMODEL + h * DHEAD;

  for (int c = 0; c < SEQ / KCHUNK; ++c) {
    __syncthreads();
    const bf16* ksrc = kbaseg + (size_t)(c * KCHUNK + kk) * DMODEL + dg;
    const bf16* vsrc = vbaseg + (size_t)(c * KCHUNK + kk) * DMODEL + dg;
    *(float4*)&sK[kk * 32 + dg] = *(const float4*)ksrc;     // 16B copy
    V8B vv; vv.f4 = *(const float4*)vsrc;
#pragma unroll
    for (int j = 0; j < 8; ++j) sVt[(dg + j) * KCHUNK + kk] = vv.e[j];
    if (c + 1 < SEQ / KCHUNK) {                             // global_prefetch_b8
      __builtin_prefetch(ksrc + KCHUNK * DMODEL, 0, 1);
      __builtin_prefetch(vsrc + KCHUNK * DMODEL, 0, 1);
    }
    __syncthreads();

    // ---- QK^T: 4 x (16 queries x 16 keys), K-dim = DH = 32 ----
    v8f s[4];
    v8f z = {};
#pragma unroll
    for (int j = 0; j < 4; ++j) {
      AB16 kb;
      const bf16* kl = &sK[(j * 16 + r) * 32 + 8 * g];
      kb.f4[0] = *(const float4*)(kl);
      kb.f4[1] = *(const float4*)(kl + 16);
      s[j] = __builtin_amdgcn_wmma_f32_16x16x32_bf16(false, qa.v, false, kb.v,
                                                     (short)0, z, false, false);
    }

    // ---- online softmax: one permlane max tree + one sum tree per chunk ----
    float p[4][8];
#pragma unroll
    for (int i = 0; i < 8; ++i) {
      float cm = fmaxf(fmaxf(s[0][i], s[1][i]), fmaxf(s[2][i], s[3][i]));
      cm = rowmax16(cm);
      float mn = fmaxf(mrow[i], cm);
      float a  = __expf(mrow[i] - mn);
      float rs = 0.0f;
#pragma unroll
      for (int j = 0; j < 4; ++j) { p[j][i] = __expf(s[j][i] - mn); rs += p[j][i]; }
      rs = rowsum16(rs);
      lrow[i] = lrow[i] * a + rs;
      mrow[i] = mn;
      acc0[i] *= a;
      acc1[i] *= a;
    }

    // ---- C-layout -> A-layout bounce through per-wave LDS (wave-internal) ----
    bf16* pw = &sP[w][0];
#pragma unroll
    for (int i = 0; i < 8; ++i) {
      int rr = i + 8 * g;
#pragma unroll
      for (int j = 0; j < 4; ++j) pw[rr * KCHUNK + j * 16 + r] = f2bf(p[j][i]);
    }
    AB16 pa0, pa1;
    const bf16* prow = pw + r * KCHUNK + 8 * g;
    pa0.f4[0] = *(const float4*)(prow);         // keys  0..31
    pa0.f4[1] = *(const float4*)(prow + 16);
    pa1.f4[0] = *(const float4*)(prow + 32);    // keys 32..63
    pa1.f4[1] = *(const float4*)(prow + 48);

    // ---- PV: B frags from transposed V; 2 key-halves x 2 dim-halves ----
#pragma unroll
    for (int kc = 0; kc < 2; ++kc) {
      AB16 vb0, vb1;
      const bf16* vl0 = &sVt[r * KCHUNK + kc * 32 + 8 * g];          // dims 0..15
      vb0.f4[0] = *(const float4*)(vl0);
      vb0.f4[1] = *(const float4*)(vl0 + 16);
      const bf16* vl1 = &sVt[(16 + r) * KCHUNK + kc * 32 + 8 * g];   // dims 16..31
      vb1.f4[0] = *(const float4*)(vl1);
      vb1.f4[1] = *(const float4*)(vl1 + 16);
      const v16bf& pfrag = kc ? pa1.v : pa0.v;
      acc0 = __builtin_amdgcn_wmma_f32_16x16x32_bf16(false, pfrag, false, vb0.v,
                                                     (short)0, acc0, false, false);
      acc1 = __builtin_amdgcn_wmma_f32_16x16x32_bf16(false, pfrag, false, vb1.v,
                                                     (short)0, acc1, false, false);
    }
  }

  // ---- normalize + store O (bf16, feeds the O-proj GEMM) ----
#pragma unroll
  for (int i = 0; i < 8; ++i) {
    int m = q0 + i + 8 * g;
    float inv = 1.0f / lrow[i];
    bf16* dst = O + (rowbase + m) * DMODEL + h * DHEAD;
    dst[r]      = f2bf(acc0[i] * inv);
    dst[16 + r] = f2bf(acc1[i] * inv);
  }
}

// ---------------------------------------------------------------------------
// Minkowski LayerNorm: independent LN over dims [0,32) and [32,128).
// One wave per row; lane holds dims {lane, lane+32, lane+64, lane+96}.
// ---------------------------------------------------------------------------
__global__ void mink_ln(const float* __restrict__ X,
                        const float* __restrict__ gt, const float* __restrict__ bt,
                        const float* __restrict__ gs, const float* __restrict__ bs,
                        float* __restrict__ Yf, bf16* __restrict__ Yb)
{
  int lane = threadIdx.x & 31;
  int row  = blockIdx.x * (blockDim.x >> 5) + (threadIdx.x >> 5);
  const float* xr = X + (size_t)row * DMODEL;
  float v[4];
#pragma unroll
  for (int i = 0; i < 4; ++i) v[i] = xr[lane + 32 * i];

  // time-like block: dims 0..31 (one element per lane)
  float s = v[0], q = v[0] * v[0];
  for (int off = 1; off < 32; off <<= 1) { s += __shfl_xor(s, off); q += __shfl_xor(q, off); }
  float mu  = s * (1.0f / 32.0f);
  float var = q * (1.0f / 32.0f) - mu * mu;
  float rsi = rsqrtf(var + 1e-5f);
  float o0  = (v[0] - mu) * rsi * gt[lane] + bt[lane];

  // space-like block: dims 32..127 (three elements per lane)
  float ss = v[1] + v[2] + v[3];
  float qq = v[1] * v[1] + v[2] * v[2] + v[3] * v[3];
  for (int off = 1; off < 32; off <<= 1) { ss += __shfl_xor(ss, off); qq += __shfl_xor(qq, off); }
  float mus  = ss * (1.0f / 96.0f);
  float vars = qq * (1.0f / 96.0f) - mus * mus;
  float rss  = rsqrtf(vars + 1e-5f);

  float* yr = Yf + (size_t)row * DMODEL;
  yr[lane] = o0;
  if (Yb) Yb[(size_t)row * DMODEL + lane] = f2bf(o0);
#pragma unroll
  for (int i = 1; i < 4; ++i) {
    int d = lane + 32 * i;
    float o = (v[i] - mus) * rss * gs[d - 32] + bs[d - 32];
    yr[d] = o;
    if (Yb) Yb[(size_t)row * DMODEL + d] = f2bf(o);
  }
}

// ---------------------------------------------------------------------------
// Conversion helpers
// ---------------------------------------------------------------------------
__global__ void cvt_bf16_kernel(const float* __restrict__ in, bf16* __restrict__ out, int n)
{
  int i = blockIdx.x * blockDim.x + threadIdx.x;
  if (i < n) out[i] = f2bf(in[i]);
}

// W[K,N] f32 -> Wt[N,K] bf16; qfold folds eta * (1/sqrt(DH)) into Wq columns
__global__ void transpose_cvt_kernel(const float* __restrict__ W, bf16* __restrict__ Wt,
                                     int K, int N, int qfold)
{
  int idx = blockIdx.x * blockDim.x + threadIdx.x;
  if (idx >= K * N) return;
  int i = idx / N, j = idx % N;
  float sc = 1.0f;
  if (qfold) sc = ((j & 31) < 8 ? -0.17677669529f : 0.17677669529f);
  Wt[(size_t)j * K + i] = f2bf(W[idx] * sc);
}

__global__ void scale_bias_q_kernel(const float* __restrict__ bq, float* __restrict__ out)
{
  int j = threadIdx.x;
  if (j < DMODEL) out[j] = bq[j] * ((j & 31) < 8 ? -0.17677669529f : 0.17677669529f);
}

// ---------------------------------------------------------------------------
// Host orchestration
// ---------------------------------------------------------------------------
extern "C" void kernel_launch(void* const* d_in, const int* in_sizes, int n_in,
                              void* d_out, int out_size, void* d_ws, size_t ws_size,
                              hipStream_t stream)
{
  (void)in_sizes; (void)n_in; (void)out_size; (void)ws_size;

  const float* x   = (const float*)d_in[0];
  const float* Wq  = (const float*)d_in[1];
  const float* bq  = (const float*)d_in[2];
  const float* Wk  = (const float*)d_in[3];
  const float* bk  = (const float*)d_in[4];
  const float* Wv  = (const float*)d_in[5];
  const float* bv  = (const float*)d_in[6];
  const float* Wo  = (const float*)d_in[7];
  const float* bo  = (const float*)d_in[8];
  const float* g1t = (const float*)d_in[9];
  const float* b1t = (const float*)d_in[10];
  const float* g1s = (const float*)d_in[11];
  const float* b1s = (const float*)d_in[12];
  const float* W1  = (const float*)d_in[13];
  const float* bf1 = (const float*)d_in[14];
  const float* W2  = (const float*)d_in[15];
  const float* bf2 = (const float*)d_in[16];
  const float* g2t = (const float*)d_in[17];
  const float* b2t = (const float*)d_in[18];
  const float* g2s = (const float*)d_in[19];
  const float* b2s = (const float*)d_in[20];

  char* p = (char*)d_ws;
  auto alloc = [&](size_t bytes) { void* r = (void*)p; p += (bytes + 255) & ~(size_t)255; return r; };

  bf16* xbf  = (bf16*)alloc((size_t)MTOK * DMODEL * 2);
  bf16* WqT  = (bf16*)alloc((size_t)DMODEL * DMODEL * 2);
  bf16* WkT  = (bf16*)alloc((size_t)DMODEL * DMODEL * 2);
  bf16* WvT  = (bf16*)alloc((size_t)DMODEL * DMODEL * 2);
  bf16* WoT  = (bf16*)alloc((size_t)DMODEL * DMODEL * 2);
  bf16* W1T  = (bf16*)alloc((size_t)DMODEL * DFF * 2);
  bf16* W2T  = (bf16*)alloc((size_t)DFF * DMODEL * 2);
  float* bqs = (float*)alloc(DMODEL * 4);
  bf16* Qb   = (bf16*)alloc((size_t)MTOK * DMODEL * 2);
  bf16* Kb   = (bf16*)alloc((size_t)MTOK * DMODEL * 2);
  bf16* Vb   = (bf16*)alloc((size_t)MTOK * DMODEL * 2);
  bf16* Ob   = (bf16*)alloc((size_t)MTOK * DMODEL * 2);
  float* y1  = (float*)alloc((size_t)MTOK * DMODEL * 4);
  float* x1  = (float*)alloc((size_t)MTOK * DMODEL * 4);
  bf16* x1b  = (bf16*)alloc((size_t)MTOK * DMODEL * 2);
  bf16* hb   = (bf16*)alloc((size_t)MTOK * DFF * 2);
  float* y2  = (float*)alloc((size_t)MTOK * DMODEL * 4);

  // 1) conversions
  {
    int n = MTOK * DMODEL;
    cvt_bf16_kernel<<<(n + 255) / 256, 256, 0, stream>>>(x, xbf, n);
    int nw = DMODEL * DMODEL;
    transpose_cvt_kernel<<<(nw + 255) / 256, 256, 0, stream>>>(Wq, WqT, DMODEL, DMODEL, 1);
    transpose_cvt_kernel<<<(nw + 255) / 256, 256, 0, stream>>>(Wk, WkT, DMODEL, DMODEL, 0);
    transpose_cvt_kernel<<<(nw + 255) / 256, 256, 0, stream>>>(Wv, WvT, DMODEL, DMODEL, 0);
    transpose_cvt_kernel<<<(nw + 255) / 256, 256, 0, stream>>>(Wo, WoT, DMODEL, DMODEL, 0);
    int n1 = DMODEL * DFF;
    transpose_cvt_kernel<<<(n1 + 255) / 256, 256, 0, stream>>>(W1, W1T, DMODEL, DFF, 0);
    transpose_cvt_kernel<<<(n1 + 255) / 256, 256, 0, stream>>>(W2, W2T, DFF, DMODEL, 0);
    scale_bias_q_kernel<<<1, 128, 0, stream>>>(bq, bqs);
  }

  auto gemm_blocks = [](int M, int N) { return ((M / 16) * (N / 16)) / 8; };

  // 2) QKV projections (Q gets eta/scale-folded weights & bias)
  gemm16<0><<<gemm_blocks(MTOK, DMODEL), 256, 0, stream>>>(xbf, WqT, bqs, nullptr, Qb, MTOK, DMODEL, DMODEL);
  gemm16<0><<<gemm_blocks(MTOK, DMODEL), 256, 0, stream>>>(xbf, WkT, bk,  nullptr, Kb, MTOK, DMODEL, DMODEL);
  gemm16<0><<<gemm_blocks(MTOK, DMODEL), 256, 0, stream>>>(xbf, WvT, bv,  nullptr, Vb, MTOK, DMODEL, DMODEL);

  // 3) flash attention: grid = B * H * (S / 128)
  lorentz_flash_attn<<<BATCH * NHEAD * (SEQ / 128), 256, 0, stream>>>(Qb, Kb, Vb, Ob);

  // 4) O-proj + residual(x) -> y1 (f32)
  gemm16<1><<<gemm_blocks(MTOK, DMODEL), 256, 0, stream>>>(Ob, WoT, bo, x, y1, MTOK, DMODEL, DMODEL);

  // 5) Minkowski LN #1 -> x1 (f32) + x1b (bf16)
  mink_ln<<<MTOK / 8, 256, 0, stream>>>(y1, g1t, b1t, g1s, b1s, x1, x1b);

  // 6) FFN up + exact GELU -> hb (bf16)
  gemm16<2><<<gemm_blocks(MTOK, DFF), 256, 0, stream>>>(x1b, W1T, bf1, nullptr, hb, MTOK, DFF, DMODEL);

  // 7) FFN down + residual(x1) -> y2 (f32)
  gemm16<1><<<gemm_blocks(MTOK, DMODEL), 256, 0, stream>>>(hb, W2T, bf2, x1, y2, MTOK, DMODEL, DFF);

  // 8) Minkowski LN #2 -> d_out (f32)
  mink_ln<<<MTOK / 8, 256, 0, stream>>>(y2, g2t, b2t, g2s, b2s, (float*)d_out, nullptr);
}
